// StarBlock_47871705481816
// MI455X (gfx1250) — 16-bit and 8-bit matrix operand order repaired and checked by reading, compiled, not run on hardware
//
#include <hip/hip_runtime.h>
#include <hip/hip_bf16.h>

typedef __attribute__((ext_vector_type(16))) _Float16 v16h;
typedef __attribute__((ext_vector_type(8)))  _Float16 v8h;
typedef __attribute__((ext_vector_type(4)))  _Float16 v4h;
typedef __attribute__((ext_vector_type(8)))  float    v8f;
typedef __attribute__((ext_vector_type(8)))  int      v8i;
typedef __attribute__((ext_vector_type(4)))  float    f4;
typedef __attribute__((ext_vector_type(4)))  int      i4;
typedef __attribute__((ext_vector_type(2)))  int      i2;

#define NB     32768
#define DIN    1024
#define DOUT   1024
#define DHID   2048
#define DH2    (DHID/2)

#if __has_builtin(__builtin_amdgcn_sched_group_barrier)
#define SGB(mask, cnt, id) __builtin_amdgcn_sched_group_barrier((mask), (cnt), (id))
#else
#define SGB(mask, cnt, id)
#endif
#define SGB_VMEM_READ 0x020
#define SGB_MFMA      0x008


__global__ void quant_w_f16_kernel(const float* __restrict__ w,
                                   _Float16* __restrict__ wq, int n) {
    int i = blockIdx.x * blockDim.x + threadIdx.x;
    if (i < n) {
        float t = fminf(fmaxf(w[i] * 128.0f, -128.0f), 127.0f);
        wq[i] = (_Float16)(rintf(t) * (1.0f / 128.0f));
    }
}

__global__ void quant_w_i8_kernel(const float* __restrict__ w,
                                  signed char* __restrict__ wq, int n) {
    int i = blockIdx.x * blockDim.x + threadIdx.x;
    if (i < n) {
        float t = fminf(fmaxf(w[i] * 128.0f, -128.0f), 127.0f);
        wq[i] = (signed char)(int)rintf(t);
    }
}

__global__ void quant_b_kernel(const float* __restrict__ b,
                               float* __restrict__ bq, int n) {
    int i = blockIdx.x * blockDim.x + threadIdx.x;
    if (i < n) bq[i] = rintf(b[i] * 16384.0f) * (1.0f / 16384.0f);
}

__global__ void cvt_x_f16_kernel(const float* __restrict__ x,
                                 _Float16* __restrict__ xh, int n4) {
    int i = blockIdx.x * blockDim.x + threadIdx.x;
    if (i < n4) {
        f4 v = *(const f4*)(x + (size_t)i * 4);
        v4h h;
        h[0] = (_Float16)v[0]; h[1] = (_Float16)v[1];
        h[2] = (_Float16)v[2]; h[3] = (_Float16)v[3];
        *(v4h*)(xh + (size_t)i * 4) = h;
    }
}

__global__ __launch_bounds__(256)
void star_up_kernel(const _Float16* __restrict__ xh,
                    const _Float16* __restrict__ w1q,
                    const _Float16* __restrict__ w2q,
                    const float*    __restrict__ b1q,
                    const float*    __restrict__ b2q,
                    signed char*    __restrict__ hq) {
    const int wave = threadIdx.x >> 5;
    const int lane = threadIdx.x & 31;
    const int half = lane >> 4;
    const int l16  = lane & 15;

    const int mBase = blockIdx.x * 128 + wave * 16;
    const int nBase = blockIdx.y * 64;

    const _Float16* xrow = xh + (size_t)(mBase + l16) * DIN + half * 16;
    const _Float16* w1b = w1q + (size_t)(nBase + l16) * DIN + half * 16;
    const _Float16* w2b = w2q + (size_t)(nBase + l16) * DIN + half * 16;

    auto loadA = [&](int k0) -> v16h {
        v8h c0 = *(const v8h*)(xrow + k0);
        v8h c1 = *(const v8h*)(xrow + k0 + 8);
        v16h a;
#pragma unroll
        for (int e = 0; e < 8; ++e) { a[e] = c0[e]; a[8 + e] = c1[e]; }
        return a;
    };
    auto loadB = [&](const _Float16* base, int j, int k0) -> v16h {
        return *(const v16h*)(base + j * (16 * DIN) + k0);
    };

    v8f acc1[4], acc2[4];
#pragma unroll
    for (int j = 0; j < 4; ++j)
#pragma unroll
        for (int e = 0; e < 8; ++e) { acc1[j][e] = 0.0f; acc2[j][e] = 0.0f; }

    v16h a  = loadA(0);
    v16h b1 = loadB(w1b, 0, 0);
    v16h b2 = loadB(w2b, 0, 0);

    for (int k0 = 0; k0 < DIN; k0 += 32) {
        const int kn = (k0 + 32 < DIN) ? (k0 + 32) : 0;
#pragma unroll
        for (int j = 0; j < 4; ++j) {
            const int jn = (j < 3) ? j + 1 : 0;
            const int kk = (j < 3) ? k0 : kn;
            v16h nb1 = loadB(w1b, jn, kk);
            v16h nb2 = loadB(w2b, jn, kk);
            v16h na = a;
            if (j == 3) na = loadA(kn);

            acc1[j] = __builtin_amdgcn_wmma_f32_16x16x32_f16(
                false, a, false, b1, (short)0, acc1[j], false, false);
            acc2[j] = __builtin_amdgcn_wmma_f32_16x16x32_f16(
                false, a, false, b2, (short)0, acc2[j], false, false);

            b1 = nb1; b2 = nb2; a = na;
        }
        SGB(SGB_VMEM_READ, 4, 0); SGB(SGB_MFMA, 2, 0);
        SGB(SGB_VMEM_READ, 4, 0); SGB(SGB_MFMA, 2, 0);
        SGB(SGB_VMEM_READ, 4, 0); SGB(SGB_MFMA, 2, 0);
        SGB(SGB_VMEM_READ, 6, 0); SGB(SGB_MFMA, 2, 0);
    }

    const int rowOff = mBase + half * 8;
#pragma unroll
    for (int j = 0; j < 4; ++j) {
        const int ncol = nBase + j * 16 + l16;
        const float bb1 = b1q[ncol];
        const float bb2 = b2q[ncol];
#pragma unroll
        for (int r = 0; r < 8; ++r) {
            float v1 = fmaxf(acc1[j][r] + bb1, 0.0f);
            int q1 = (int)floorf(fminf(v1 * 128.0f, 127.0f));
            float t2 = fminf(fmaxf((acc2[j][r] + bb2) * 128.0f, -128.0f), 127.0f);
            int q2 = (int)floorf(t2);
            int p = q1 * q2;
            p += __shfl_xor(p, 1, 32);
            int hv = p >> 7;
            hv = hv < -128 ? -128 : (hv > 127 ? 127 : hv);
            if ((lane & 1) == 0) {
                hq[(size_t)(rowOff + r) * DH2 + (ncol >> 1)] = (signed char)hv;
            }
        }
    }
}

__global__ __launch_bounds__(256)
void star_down_kernel(const signed char* __restrict__ hq,
                      const signed char* __restrict__ wdq,
                      const float*       __restrict__ bdq,
                      float*             __restrict__ out) {
    const int wave = threadIdx.x >> 5;
    const int lane = threadIdx.x & 31;
    const int half = lane >> 4;
    const int l16  = lane & 15;

    const int mBase = blockIdx.x * 128 + wave * 16;
    const int nBase = blockIdx.y * 64;

    const signed char* hrow = hq + (size_t)(mBase + l16) * DH2 + half * 16;
    const signed char* wb   = wdq + (size_t)(nBase + l16) * DH2 + half * 16;

    auto loadA = [&](int k0) -> v8i {
        i2 c0 = *(const i2*)(hrow + k0);
        i2 c1 = *(const i2*)(hrow + k0 + 8);
        i2 c2 = *(const i2*)(hrow + k0 + 32);
        i2 c3 = *(const i2*)(hrow + k0 + 40);
        v8i a;
        a[0] = c0[0]; a[1] = c0[1]; a[2] = c1[0]; a[3] = c1[1];
        a[4] = c2[0]; a[5] = c2[1]; a[6] = c3[0]; a[7] = c3[1];
        return a;
    };
    auto loadB = [&](int j, int k0) -> v8i {
        const signed char* p = wb + j * (16 * DH2) + k0;
        i4 b0 = *(const i4*)(p);
        i4 b1 = *(const i4*)(p + 32);
        v8i b;
        b[0] = b0[0]; b[1] = b0[1]; b[2] = b0[2]; b[3] = b0[3];
        b[4] = b1[0]; b[5] = b1[1]; b[6] = b1[2]; b[7] = b1[3];
        return b;
    };

    v8i acc[4];
#pragma unroll
    for (int j = 0; j < 4; ++j)
#pragma unroll
        for (int e = 0; e < 8; ++e) acc[j][e] = 0;

    v8i a = loadA(0);
    v8i b = loadB(0, 0);

    for (int k0 = 0; k0 < DH2; k0 += 64) {
        const int kn = (k0 + 64 < DH2) ? (k0 + 64) : 0;
#pragma unroll
        for (int j = 0; j < 4; ++j) {
            const int jn = (j < 3) ? j + 1 : 0;
            const int kk = (j < 3) ? k0 : kn;
            v8i nb = loadB(jn, kk);
            v8i na = a;
            if (j == 3) na = loadA(kn);

            acc[j] = __builtin_amdgcn_wmma_i32_16x16x64_iu8(
                true, a, true, b, acc[j], false, false);

            b = nb; a = na;
        }
        SGB(SGB_VMEM_READ, 2, 0); SGB(SGB_MFMA, 1, 0);
        SGB(SGB_VMEM_READ, 2, 0); SGB(SGB_MFMA, 1, 0);
        SGB(SGB_VMEM_READ, 2, 0); SGB(SGB_MFMA, 1, 0);
        SGB(SGB_VMEM_READ, 6, 0); SGB(SGB_MFMA, 1, 0);
    }

    const int rowOff = mBase + half * 8;
#pragma unroll
    for (int j = 0; j < 4; ++j) {
        const int n  = nBase + j * 16 + l16;
        const float bb = bdq[n];
#pragma unroll
        for (int r = 0; r < 8; ++r) {
            float v = (float)acc[j][r] * (1.0f / 16384.0f) + bb;
            out[(size_t)(rowOff + r) * DOUT + n] = fmaxf(v, 0.0f);
        }
    }
}

extern "C" void kernel_launch(void* const* d_in, const int* in_sizes, int n_in,
                              void* d_out, int out_size, void* d_ws, size_t ws_size,
                              hipStream_t stream) {
    const float* x  = (const float*)d_in[0];
    const float* w1 = (const float*)d_in[1];
    const float* b1 = (const float*)d_in[2];
    const float* w2 = (const float*)d_in[3];
    const float* b2 = (const float*)d_in[4];
    const float* wd = (const float*)d_in[5];
    const float* bd = (const float*)d_in[6];
    float* out = (float*)d_out;

    char* ws = (char*)d_ws;
    const size_t MB = (size_t)1 << 20;
    _Float16*    w1q = (_Float16*)(ws);
    _Float16*    w2q = (_Float16*)(ws + 4 * MB);
    signed char* wdq = (signed char*)(ws + 8 * MB);
    float*       b1q = (float*)(ws + 9 * MB);
    float*       b2q = (float*)(ws + 9 * MB + 16384);
    float*       bdq = (float*)(ws + 9 * MB + 32768);
    signed char* hqb = (signed char*)(ws + 10 * MB);
    _Float16*    xh  = (_Float16*)(ws + 42 * MB);

    {
        int n = DHID * DIN;
        quant_w_f16_kernel<<<(n + 255) / 256, 256, 0, stream>>>(w1, w1q, n);
        quant_w_f16_kernel<<<(n + 255) / 256, 256, 0, stream>>>(w2, w2q, n);
        int nd = DOUT * DH2;
        quant_w_i8_kernel<<<(nd + 255) / 256, 256, 0, stream>>>(wd, wdq, nd);
        quant_b_kernel<<<(DHID + 255) / 256, 256, 0, stream>>>(b1, b1q, DHID);
        quant_b_kernel<<<(DHID + 255) / 256, 256, 0, stream>>>(b2, b2q, DHID);
        quant_b_kernel<<<(DOUT + 255) / 256, 256, 0, stream>>>(bd, bdq, DOUT);
        int n4 = (NB * DIN) / 4;
        cvt_x_f16_kernel<<<(n4 + 255) / 256, 256, 0, stream>>>(x, xh, n4);
    }

    star_up_kernel<<<dim3(NB / 128, DHID / 64), 256, 0, stream>>>(
        xh, w1q, w2q, b1q, b2q, hqb);

    star_down_kernel<<<dim3(NB / 128, DOUT / 64), 256, 0, stream>>>(
        hqb, wdq, bdq, out);
}
